// YakirCorrelationCross_55886114455786
// MI455X (gfx1250) — matrix operand order repaired and checked by reading, not compiled, not run on hardware
//
#include <hip/hip_runtime.h>
#include <hip/hip_bf16.h>
#include <math.h>

// ---------------------------------------------------------------------------
// FEDformer-style wavelet/Fourier attention for MI455X (gfx1250, wave32).
// Heavy parts (17-tap 128x128 convs, 128x128 projections) -> WMMA bf16.
// Activation staging: Tensor Data Mover (tensor_load_to_lds) into LDS.
// Cheap spectral parts (64-mode DFT / mode-mix / iDFT / 16-mode cross-attn)
// -> f32 VALU with incremental rotations.
// ---------------------------------------------------------------------------

#define DEV __device__ __forceinline__

typedef __attribute__((ext_vector_type(16))) __bf16 v16bf;
typedef __attribute__((ext_vector_type(8)))  float  v8f;
typedef unsigned short ushort_t;

#ifndef PI_F
#define PI_F 3.14159265358979323846f
#endif

DEV float bf2f(ushort_t u) {
    unsigned v = ((unsigned)u) << 16;
    return __builtin_bit_cast(float, v);
}
DEV ushort_t f2bf(float f) {
    unsigned u = __builtin_bit_cast(unsigned, f);
    unsigned r = (u + 0x7FFFu + ((u >> 16) & 1u)) >> 16;   // RNE
    return (ushort_t)r;
}
DEV __bf16 us2bf(ushort_t u) { return __builtin_bit_cast(__bf16, u); }

DEV v8f wmma_bf16(v16bf a, v16bf b, v8f c) {
#if defined(__HIP_DEVICE_COMPILE__)
    // v_wmma_f32_16x16x32_bf16 : D = A(16x32) x B(32x16) + C
    return __builtin_amdgcn_wmma_f32_16x16x32_bf16(false, a, false, b,
                                                   (short)0, c, false, false);
#else
    (void)a; (void)b;
    return c;
#endif
}

// ---------------------------------------------------------------------------
// Tensor Data Mover: 2D tile (tile_h rows x tile_w bf16 elems) global -> LDS.
// Descriptor packing per CDNA5 ISA (D# group0/group1).
// This toolchain exposes the 6-arg builtin (g0 u32x4, g1 i32x8, g2 i32x4,
// g3 i32x4, g4 i32x8, cpol) -- groups 2/3/4 are zero for a 2D tile.
// ---------------------------------------------------------------------------
#if defined(__HIP_DEVICE_COMPILE__) && __has_builtin(__builtin_amdgcn_tensor_load_to_lds) && __has_builtin(__builtin_amdgcn_s_wait_tensorcnt)
#define HAVE_TDM 1
#else
#define HAVE_TDM 0
#endif

#if HAVE_TDM
DEV void tdm_load_2d(unsigned lds_off, const void* g, int tile_w, int tile_h,
                     int row_pitch_elems) {
    typedef unsigned int u32x4v __attribute__((ext_vector_type(4)));
    typedef int          i32x8v __attribute__((ext_vector_type(8)));
    typedef int          i32x4v __attribute__((ext_vector_type(4)));
    unsigned long long ga = (unsigned long long)g;
    u32x4v g0;
    g0[0] = 1u;                                   // count=1, user mode
    g0[1] = lds_off;                              // lds_addr (bytes)
    g0[2] = (unsigned)ga;                         // global_addr[31:0]
    g0[3] = (unsigned)((ga >> 32) & 0x01FFFFFFull) | (2u << 30); // [56:32] | type=2
    i32x8v g1;
    g1[0] = (1 << 16);                            // data_size = 1 (2 bytes/elem)
    g1[1] = (int)(((unsigned)row_pitch_elems & 0xFFFFu) << 16);         // dim0 lo
    g1[2] = (int)((((unsigned)row_pitch_elems >> 16) & 0xFFFFu)
                  | (((unsigned)tile_h & 0xFFFFu) << 16));              // dim0 hi | dim1 lo
    g1[3] = (int)(((unsigned)tile_w & 0xFFFFu) << 16);                  // tile_dim0
    g1[4] = (int)((unsigned)tile_h & 0xFFFFu);                          // tile_dim1
    g1[5] = (int)(unsigned)row_pitch_elems;                             // dim0_stride lo
    g1[6] = 0;
    g1[7] = 0;
    i32x4v z4; z4[0] = 0; z4[1] = 0; z4[2] = 0; z4[3] = 0;
    i32x8v z8; z8[0] = 0; z8[1] = 0; z8[2] = 0; z8[3] = 0;
    z8[4] = 0; z8[5] = 0; z8[6] = 0; z8[7] = 0;
    __builtin_amdgcn_tensor_load_to_lds(g0, g1, z4, z4, z8, 0);
}
#endif

// ---------------------------------------------------------------------------
// Generic WMMA conv1d kernel (also does the final 1x1 GEMM projection).
//   out[b, co, t] = sum_p sum_ci sum_f w_p[co,ci,f] * x_p[b, ci, stride*t+f-pad]
//                   + bias
// Inputs are bf16, padded layout (b, 128, nin+32), halo of 16 zeros each side.
// Block: 256 threads = 8 waves; wave w owns co in [16w,16w+16), 32 time steps.
// ---------------------------------------------------------------------------
#define CONV_T 32
#define XW_MAX 80   // stride2 tile width: 2*32+16 ; stride1 uses 48

__global__ __launch_bounds__(256) void conv_wmma_kernel(
    const ushort_t* __restrict__ x1, const float* __restrict__ w1,
    const float* __restrict__ bb1,
    const ushort_t* __restrict__ x2, const float* __restrict__ w2,
    const float* __restrict__ bb2,
    ushort_t* __restrict__ ybf, float* __restrict__ yf32,
    float* __restrict__ stats,
    int nin, int nout, int stride, int taps, int pad)
{
    __shared__ ushort_t xl[128 * XW_MAX];
    __shared__ ushort_t wl[128 * 128];

    const int tid  = threadIdx.x;
    const int lane = tid & 31;
    const int wv   = tid >> 5;
    const int co0  = wv * 16;
    const int bi   = blockIdx.y;
    const int t0   = blockIdx.x * CONV_T;
    const int ipitch = nin + 32;
    const int width  = stride * CONV_T + 16;
    const int aoff   = 8 - pad;
    const int npaths = (x2 != nullptr) ? 2 : 1;

    v8f acc0 = {}; v8f acc1 = {};

    for (int p = 0; p < npaths; ++p) {
        const ushort_t* xin = p ? x2 : x1;
        const float*    win = p ? w2 : w1;
        __syncthreads();
        const ushort_t* gsrc = xin + (size_t)(bi * 128) * ipitch
                                   + (16 + stride * t0 - 8);
#if HAVE_TDM
        if (wv == 0) {
            unsigned lds_off = (unsigned)(unsigned long long)(void*)&xl[0];
            tdm_load_2d(lds_off, (const void*)gsrc, width, 128, ipitch);
            __builtin_amdgcn_s_wait_tensorcnt(0);
        }
        __syncthreads();
#else
        {
            int lim = ipitch - (16 + stride * t0 - 8);
            for (int i = tid; i < 128 * width; i += 256) {
                int ci = i / width, c = i - ci * width;
                xl[ci * width + c] = (c < lim) ? gsrc[(size_t)ci * ipitch + c]
                                               : (ushort_t)0;
            }
        }
        __syncthreads();
#endif
        __builtin_prefetch(win, 0, 1);
        for (int f = 0; f < taps; ++f) {
            __syncthreads();
            for (int i = tid; i < 128 * 128; i += 256)
                wl[i] = f2bf(win[(size_t)i * taps + f]);
            __syncthreads();

#pragma unroll
            for (int ks = 0; ks < 4; ++ks) {
                v16bf bfr;
                {
                    int grp = lane >> 4, nn = lane & 15;
                    const ushort_t* col = &wl[(co0 + nn) * 128 + ks * 32 + grp * 8];
#pragma unroll
                    for (int v = 0; v < 8; ++v) {
                        int o = (v < 4) ? 2 * v : 2 * v + 8;
                        bfr[2 * v]     = us2bf(col[o]);
                        bfr[2 * v + 1] = us2bf(col[o + 1]);
                    }
                }
#pragma unroll
                for (int mt = 0; mt < 2; ++mt) {
                    v16bf afr;
                    int half = lane >> 4, m = lane & 15;
                    int c = aoff + f + stride * (mt * 16 + m);
#pragma unroll
                    for (int v = 0; v < 8; ++v) {
                        int k = ks * 32 + ((v < 4) ? (half * 8 + 2 * v)
                                                   : (16 + half * 8 + 2 * (v - 4)));
                        afr[2 * v]     = us2bf(xl[k * width + c]);
                        afr[2 * v + 1] = us2bf(xl[(k + 1) * width + c]);
                    }
                    if (mt == 0) acc0 = wmma_bf16(afr, bfr, acc0);
                    else         acc1 = wmma_bf16(afr, bfr, acc1);
                }
            }
        }
    }

    // epilogue: C/D layout -> lane col = lane&15 (co), row = (lane>>4)*8 + r (t)
    int co = co0 + (lane & 15);
    float bias = bb1 ? bb1[co] : 0.f;
    if (npaths == 2 && bb2) bias += bb2[co];
    float s1 = 0.f, s2 = 0.f;
    const int opitch = nout + 32;
#pragma unroll
    for (int mt = 0; mt < 2; ++mt) {
        v8f acc = mt ? acc1 : acc0;
#pragma unroll
        for (int r = 0; r < 8; ++r) {
            int t = t0 + mt * 16 + (lane >> 4) * 8 + r;
            float v = acc[r] + bias;
            if (t < nout) {
                s1 += v; s2 += v * v;
                if (ybf)
                    ybf[(size_t)(bi * 128 + co) * opitch + 16 + t] = f2bf(v);
                if (yf32)
                    yf32[((size_t)bi * nout + t) * 128 + co] = v;
            }
        }
    }
    if (stats) {
        for (int o = 16; o > 0; o >>= 1) {
            s1 += __shfl_xor(s1, o, 32);
            s2 += __shfl_xor(s2, o, 32);
        }
        if (lane == 0) { atomicAdd(&stats[0], s1); atomicAdd(&stats[1], s2); }
    }
}

// ---------------------------------------------------------------------------
// Projection: y[b, co, t] = sum_d x[b, t, d] * W[co, d] + bias[co]
// x is f32 (B,N,128); y is bf16 padded (B,128,N+32).
// ---------------------------------------------------------------------------
__global__ __launch_bounds__(256) void proj_wmma_kernel(
    const float* __restrict__ x, const float* __restrict__ w,
    const float* __restrict__ bb, ushort_t* __restrict__ y, int N_)
{
    __shared__ ushort_t xl[32 * 128];
    __shared__ ushort_t wl[128 * 128];
    const int tid = threadIdx.x, lane = tid & 31, wv = tid >> 5, co0 = wv * 16;
    const int bi = blockIdx.y, t0 = blockIdx.x * 32;

    for (int i = tid; i < 128 * 128; i += 256) wl[i] = f2bf(w[i]);
    const float* xs = x + ((size_t)bi * N_ + t0) * 128;
    for (int i = tid; i < 32 * 128; i += 256) xl[i] = f2bf(xs[i]);
    __syncthreads();

    v8f acc0 = {}, acc1 = {};
#pragma unroll
    for (int ks = 0; ks < 4; ++ks) {
        v16bf bfr;
        {
            int grp = lane >> 4, nn = lane & 15;
            const ushort_t* col = &wl[(co0 + nn) * 128 + ks * 32 + grp * 8];
#pragma unroll
            for (int v = 0; v < 8; ++v) {
                int o = (v < 4) ? 2 * v : 2 * v + 8;
                bfr[2 * v]     = us2bf(col[o]);
                bfr[2 * v + 1] = us2bf(col[o + 1]);
            }
        }
#pragma unroll
        for (int mt = 0; mt < 2; ++mt) {
            v16bf afr;
            int half = lane >> 4, m = lane & 15;
            const ushort_t* row = &xl[(mt * 16 + m) * 128 + ks * 32];
#pragma unroll
            for (int v = 0; v < 8; ++v) {
                int k = (v < 4) ? (half * 8 + 2 * v) : (16 + half * 8 + 2 * (v - 4));
                afr[2 * v]     = us2bf(row[k]);
                afr[2 * v + 1] = us2bf(row[k + 1]);
            }
            if (mt == 0) acc0 = wmma_bf16(afr, bfr, acc0);
            else         acc1 = wmma_bf16(afr, bfr, acc1);
        }
    }
    int co = co0 + (lane & 15);
    float bias = bb[co];
    const int opitch = N_ + 32;
#pragma unroll
    for (int mt = 0; mt < 2; ++mt) {
        v8f a = mt ? acc1 : acc0;
#pragma unroll
        for (int r = 0; r < 8; ++r) {
            int t = t0 + mt * 16 + (lane >> 4) * 8 + r;
            y[(size_t)(bi * 128 + co) * opitch + 16 + t] = f2bf(a[r] + bias);
        }
    }
}

// ---------------------------------------------------------------------------
// Forward DFT: F[bc, m] = sum_t x[bc, t] * exp(-2 pi i m t / n), m < L <= 64.
// grid = 1024 (b*128 channels), block = 64 (one mode / thread).
// ---------------------------------------------------------------------------
__global__ __launch_bounds__(64) void dft_kernel(
    const ushort_t* __restrict__ x, float* __restrict__ F, int n, int L)
{
    int bc = blockIdx.x;
    int m  = threadIdx.x;
    if (m >= L) return;
    const ushort_t* row = x + (size_t)bc * (n + 32) + 16;
    float ang = -2.f * PI_F * (float)m / (float)n;
    float wr = cosf(ang), wi = sinf(ang), cr = 1.f, ci = 0.f;
    float ar = 0.f, ai = 0.f;
    for (int t = 0; t < n; ++t) {
        float xv = bf2f(row[t]);
        ar += xv * cr; ai += xv * ci;
        float nr = cr * wr - ci * wi;
        ci = cr * wi + ci * wr; cr = nr;
    }
    F[((size_t)bc * 64 + m) * 2]     = ar;
    F[((size_t)bc * 64 + m) * 2 + 1] = ai;
}

// ---------------------------------------------------------------------------
// Per-mode complex mixing: G[b,o,m] = sum_i F[b,i,m] * (wr+ i wi)[i,o,m].
// grid = (L, B), block = 128 (one o / thread).
// ---------------------------------------------------------------------------
__global__ __launch_bounds__(128) void modemix_kernel(
    const float* __restrict__ F, const float* __restrict__ wr,
    const float* __restrict__ wi, float* __restrict__ G, int L)
{
    __shared__ float fs[128 * 2];
    int m = blockIdx.x, b = blockIdx.y, o = threadIdx.x;
    fs[o * 2]     = F[(((size_t)b * 128 + o) * 64 + m) * 2];
    fs[o * 2 + 1] = F[(((size_t)b * 128 + o) * 64 + m) * 2 + 1];
    __syncthreads();
    float gr = 0.f, gi = 0.f;
    for (int i = 0; i < 128; ++i) {
        float ar = fs[i * 2], ai = fs[i * 2 + 1];
        float br = wr[((size_t)i * 128 + o) * 64 + m];
        float bi = wi[((size_t)i * 128 + o) * 64 + m];
        gr += ar * br - ai * bi;
        gi += ar * bi + ai * br;
    }
    G[(((size_t)b * 128 + o) * 64 + m) * 2]     = gr;
    G[(((size_t)b * 128 + o) * 64 + m) * 2 + 1] = gi;
    (void)L;
}

// ---------------------------------------------------------------------------
// Inverse DFT (truncated modes) + global sum/sumsq stats.
// ---------------------------------------------------------------------------
__global__ __launch_bounds__(256) void idft_kernel(
    const float* __restrict__ G, ushort_t* __restrict__ y,
    float* __restrict__ stats, int n, int L)
{
    __shared__ float gs[64 * 2];
    int bc = blockIdx.x;
    int t  = blockIdx.y * 256 + threadIdx.x;
    if (threadIdx.x < 64) {
        int m = threadIdx.x;
        gs[m * 2]     = (m < L) ? G[((size_t)bc * 64 + m) * 2]     : 0.f;
        gs[m * 2 + 1] = (m < L) ? G[((size_t)bc * 64 + m) * 2 + 1] : 0.f;
    }
    __syncthreads();
    float vv = 0.f;
    if (t < n) {
        float ang = 2.f * PI_F * (float)t / (float)n;
        float wr = cosf(ang), wi = sinf(ang), cr = 1.f, ci = 0.f;
        for (int m = 0; m < L; ++m) {
            float fac = (m == 0 || 2 * m == n) ? 1.f : 2.f;
            vv += fac * (gs[m * 2] * cr - gs[m * 2 + 1] * ci);
            float nr = cr * wr - ci * wi;
            ci = cr * wi + ci * wr; cr = nr;
        }
        vv /= (float)n;
        y[(size_t)bc * (n + 32) + 16 + t] = f2bf(vv);
    }
    float s1 = (t < n) ? vv : 0.f;
    float s2 = (t < n) ? vv * vv : 0.f;
    for (int o = 16; o > 0; o >>= 1) {
        s1 += __shfl_xor(s1, o, 32);
        s2 += __shfl_xor(s2, o, 32);
    }
    if ((threadIdx.x & 31) == 0) {
        atomicAdd(&stats[0], s1);
        atomicAdd(&stats[1], s2);
    }
}

// y = (y - subMean*mean) / std(ddof=1), in place over interior.
__global__ __launch_bounds__(256) void normalize_kernel(
    ushort_t* __restrict__ y, const float* __restrict__ stats,
    float cnt, int subMean, int n)
{
    int idx = blockIdx.x * 256 + threadIdx.x;
    int total = 1024 * n;
    if (idx >= total) return;
    int bc = idx / n, t = idx - bc * n;
    float mean = stats[0] / cnt;
    float var  = (stats[1] - stats[0] * stats[0] / cnt) / (cnt - 1.f);
    float inv  = rsqrtf(fmaxf(var, 1e-12f));
    size_t p = (size_t)bc * (n + 32) + 16 + t;
    float v = bf2f(y[p]);
    v = (v - (subMean ? mean : 0.f)) * inv;
    y[p] = f2bf(v);
}

// complex tanh: tanh(a+bi) = (tanh a + i tan b) / (1 + i tanh a tan b)
DEV void ctanh_c(float a, float b, float& re, float& im) {
    float th = tanhf(a), tb = tanf(b);
    float d = 1.f + th * th * tb * tb;
    re = th * (1.f + tb * tb) / d;
    im = tb * (1.f - th * th) / d;
}

// ---------------------------------------------------------------------------
// Fourier cross attention (e=1): per channel bc, 16 modes of q and k,
// A[x,y] = tanh(Fq[x]*Fk[y]), out modes om[x] = sum_y A[x,y]*Fk[y], irfft.
// ---------------------------------------------------------------------------
__global__ __launch_bounds__(64) void fca_kernel(
    const ushort_t* __restrict__ q, const ushort_t* __restrict__ k,
    ushort_t* __restrict__ out, int n)
{
    __shared__ float fq[16 * 2], fk[16 * 2], om[16 * 2];
    int bc = blockIdx.x, tid = threadIdx.x;
    int mq = n / 2; if (mq > 16) mq = 16;
    const int pitch = n + 32;
    const ushort_t* qrow = q + (size_t)bc * pitch + 16;
    const ushort_t* krow = k + (size_t)bc * pitch + 16;

    if (tid < 32) {
        int m = tid & 15;
        bool isk = tid >= 16;
        if (m < mq) {
            const ushort_t* row = isk ? krow : qrow;
            float ang = -2.f * PI_F * (float)m / (float)n;
            float wr = cosf(ang), wi = sinf(ang), cr = 1.f, ci = 0.f;
            float ar = 0.f, ai = 0.f;
            for (int t = 0; t < n; ++t) {
                float xv = bf2f(row[t]);
                ar += xv * cr; ai += xv * ci;
                float nr = cr * wr - ci * wi;
                ci = cr * wi + ci * wr; cr = nr;
            }
            if (isk) { fk[m * 2] = ar; fk[m * 2 + 1] = ai; }
            else     { fq[m * 2] = ar; fq[m * 2 + 1] = ai; }
        }
    }
    __syncthreads();
    if (tid < mq) {
        float ar = 0.f, ai = 0.f;
        float qr = fq[tid * 2], qi = fq[tid * 2 + 1];
        for (int y = 0; y < mq; ++y) {
            float kr = fk[y * 2], ki = fk[y * 2 + 1];
            float pr = qr * kr - qi * ki, pi2 = qr * ki + qi * kr;
            float tr, ti; ctanh_c(pr, pi2, tr, ti);
            ar += tr * kr - ti * ki;
            ai += tr * ki + ti * kr;
        }
        om[tid * 2] = ar; om[tid * 2 + 1] = ai;
    }
    __syncthreads();
    ushort_t* orow = out + (size_t)bc * pitch + 16;
    for (int t = tid; t < n; t += 64) {
        float vv = 0.f;
        float ang = 2.f * PI_F * (float)t / (float)n;
        float wr = cosf(ang), wi = sinf(ang), cr = 1.f, ci = 0.f;
        for (int m = 0; m < mq; ++m) {
            float fac = (m == 0) ? 1.f : 2.f;
            vv += fac * (om[m * 2] * cr - om[m * 2 + 1] * ci);
            float nr = cr * wr - ci * wi;
            ci = cr * wi + ci * wr; cr = nr;
        }
        vv /= (float)n;
        orow[t] = f2bf(vv);
    }
}

// y[bc, 2t] = y[bc, 2t+1] = x[bc, t]   (pitches n+32 -> 2n+32)
__global__ __launch_bounds__(256) void upsample_kernel(
    const ushort_t* __restrict__ x, ushort_t* __restrict__ y, int n)
{
    int idx = blockIdx.x * 256 + threadIdx.x;
    int total = 1024 * n;
    if (idx >= total) return;
    int bc = idx / n, t = idx - bc * n;
    ushort_t v = x[(size_t)bc * (n + 32) + 16 + t];
    size_t po = (size_t)bc * (2 * n + 32) + 16;
    y[po + 2 * t]     = v;
    y[po + 2 * t + 1] = v;
}

__global__ void zero2_kernel(float* p) { if (threadIdx.x < 2) p[threadIdx.x] = 0.f; }

__global__ __launch_bounds__(256) void fill_kernel(unsigned* p, size_t nwords) {
    size_t i = (size_t)blockIdx.x * 256 + threadIdx.x;
    size_t st = (size_t)gridDim.x * 256;
    for (; i < nwords; i += st) p[i] = 0u;
}

// ---------------------------------------------------------------------------
extern "C" void kernel_launch(void* const* d_in, const int* in_sizes, int n_in,
                              void* d_out, int out_size, void* d_ws, size_t ws_size,
                              hipStream_t stream)
{
    (void)in_sizes; (void)n_in; (void)out_size;
    const int Bb = 8, N = 4096, LV = 12;
    const float* xin[3] = { (const float*)d_in[0], (const float*)d_in[1],
                            (const float*)d_in[2] };
    const float* l1w[3] = { (const float*)d_in[3], (const float*)d_in[5],
                            (const float*)d_in[7] };
    const float* l1b[3] = { (const float*)d_in[4], (const float*)d_in[6],
                            (const float*)d_in[8] };
    const float* outw = (const float*)d_in[9];
    const float* outb = (const float*)d_in[10];
    // per signal: 0 hcw 1 hcb 2 lcw 3 lcb 4 hfr 5 hfi 6 lfr 7 lfi
    const float* sw[3][8];
    for (int s = 0; s < 3; ++s)
        for (int j = 0; j < 8; ++j)
            sw[s][j] = (const float*)d_in[11 + 8 * s + j];
    const float* ilpw = (const float*)d_in[35];
    const float* ilpb = (const float*)d_in[36];
    const float* ihpw = (const float*)d_in[37];
    const float* ihpb = (const float*)d_in[38];

    char* ws = (char*)d_ws;
    size_t off = 0;
    auto alloc = [&](size_t b) -> size_t {
        size_t r = off; off += (b + 255) & ~(size_t)255; return r;
    };
    auto ubuf = [&](int n) -> size_t {
        return alloc((size_t)1024 * (size_t)(n + 32) * sizeof(ushort_t));
    };
    size_t projo[3];
    for (int s = 0; s < 3; ++s) projo[s] = ubuf(N);
    size_t deto[3][LV + 1];
    for (int s = 0; s < 3; ++s)
        for (int l = 1; l <= LV; ++l) deto[s][l] = ubuf(N >> l);
    size_t pp[2] = { ubuf(N / 2), ubuf(N / 2) };
    size_t caF[3];
    for (int s = 0; s < 3; ++s) caF[s] = ubuf(1);
    size_t convtmp = ubuf(N / 2);
    size_t attno[LV + 1];
    attno[0] = ubuf(1);
    for (int l = 1; l <= LV; ++l) attno[l] = ubuf(N >> l);
    size_t upA = ubuf(N), upD = ubuf(N);
    size_t ip2[2] = { ubuf(N), ubuf(N) };
    size_t Fo = alloc((size_t)1024 * 64 * 2 * 4);
    size_t Go = alloc((size_t)1024 * 64 * 2 * 4);
    size_t stato = alloc(256);
    if (off > ws_size) return;   // insufficient scratch

    float* Fp = (float*)(ws + Fo);
    float* Gp = (float*)(ws + Go);
    float* stats = (float*)(ws + stato);

    auto cdiv = [](int a, int b) { return (a + b - 1) / b; };

    { // zero whole used workspace once (halos start as zero)
        size_t wds = off / 4;
        int blk = (int)((wds + 255) / 256); if (blk > 8192) blk = 8192;
        fill_kernel<<<dim3(blk), dim3(256), 0, stream>>>((unsigned*)ws, wds);
    }
    auto fillz = [&](size_t o, size_t bytes) {
        size_t wds = bytes / 4;
        int blk = (int)((wds + 255) / 256);
        if (blk > 8192) blk = 8192; if (blk < 1) blk = 1;
        fill_kernel<<<dim3(blk), dim3(256), 0, stream>>>((unsigned*)(ws + o), wds);
    };

    // --- linear projections q/k/v ---
    for (int s = 0; s < 3; ++s)
        proj_wmma_kernel<<<dim3(N / 32, Bb), dim3(256), 0, stream>>>(
            xin[s], l1w[s], l1b[s], (ushort_t*)(ws + projo[s]), N);

    auto conv = [&](size_t x1o, const float* w1, const float* b1,
                    bool dual, size_t x2o, const float* w2, const float* b2,
                    bool toBf, size_t ybo, float* yf, bool useStats,
                    int nin, int nout, int stride, int taps, int pad) {
        conv_wmma_kernel<<<dim3(cdiv(nout, 32), Bb), dim3(256), 0, stream>>>(
            (const ushort_t*)(ws + x1o), w1, b1,
            dual ? (const ushort_t*)(ws + x2o) : nullptr, w2, b2,
            toBf ? (ushort_t*)(ws + ybo) : nullptr, yf,
            useStats ? stats : nullptr, nin, nout, stride, taps, pad);
    };
    auto sparse = [&](size_t srco, size_t dsto, const float* mr, const float* mi,
                      int nn, int L) {
        dft_kernel<<<dim3(1024), dim3(64), 0, stream>>>(
            (const ushort_t*)(ws + srco), Fp, nn, L);
        modemix_kernel<<<dim3(L, Bb), dim3(128), 0, stream>>>(Fp, mr, mi, Gp, L);
        zero2_kernel<<<dim3(1), dim3(32), 0, stream>>>(stats);
        idft_kernel<<<dim3(1024, cdiv(nn, 256)), dim3(256), 0, stream>>>(
            Gp, (ushort_t*)(ws + dsto), stats, nn, L);
        normalize_kernel<<<dim3(cdiv(1024 * nn, 256)), dim3(256), 0, stream>>>(
            (ushort_t*)(ws + dsto), stats, (float)((size_t)1024 * nn), 1, nn);
    };

    // --- wavelet decomposition per signal ---
    for (int s = 0; s < 3; ++s) {
        size_t cur = projo[s];
        int n = N;
        for (int l = 1; l <= LV; ++l) {
            int nn = n >> 1;
            int L = nn / 2 + 1; if (L > 64) L = 64;
            // high-pass -> detail
            conv(cur, sw[s][0], sw[s][1], false, 0, nullptr, nullptr,
                 true, convtmp, nullptr, false, n, nn, 2, 17, 8);
            sparse(convtmp, deto[s][l], sw[s][4], sw[s][5], nn, L);
            // low-pass -> next approximation
            size_t dst = (l == LV) ? caF[s] : pp[l & 1];
            conv(cur, sw[s][2], sw[s][3], false, 0, nullptr, nullptr,
                 true, convtmp, nullptr, false, n, nn, 2, 17, 8);
            if (l != LV) fillz(dst, (size_t)1024 * (size_t)(nn + 32) * 2);
            sparse(convtmp, dst, sw[s][6], sw[s][7], nn, L);
            cur = dst; n = nn;
        }
    }

    // --- Fourier cross attention: cA + every detail level ---
    fca_kernel<<<dim3(1024), dim3(64), 0, stream>>>(
        (const ushort_t*)(ws + caF[0]), (const ushort_t*)(ws + caF[1]),
        (ushort_t*)(ws + attno[0]), 1);
    for (int l = 1; l <= LV; ++l)
        fca_kernel<<<dim3(1024), dim3(64), 0, stream>>>(
            (const ushort_t*)(ws + deto[0][l]), (const ushort_t*)(ws + deto[1][l]),
            (ushort_t*)(ws + attno[l]), N >> l);

    // --- inverse DWT ---
    size_t cur = attno[0];
    int pi = 0;
    for (int lv = 0; lv < LV; ++lv) {
        int ln = N >> (LV - lv);                 // current level length
        fillz(upA, (size_t)1024 * (size_t)(2 * ln + 32) * 2);
        fillz(upD, (size_t)1024 * (size_t)(2 * ln + 32) * 2);
        upsample_kernel<<<dim3(cdiv(1024 * ln, 256)), dim3(256), 0, stream>>>(
            (const ushort_t*)(ws + cur), (ushort_t*)(ws + upA), ln);
        upsample_kernel<<<dim3(cdiv(1024 * ln, 256)), dim3(256), 0, stream>>>(
            (const ushort_t*)(ws + attno[LV - lv]), (ushort_t*)(ws + upD), ln);
        fillz(ip2[pi], (size_t)1024 * (size_t)(2 * ln + 32) * 2);
        zero2_kernel<<<dim3(1), dim3(32), 0, stream>>>(stats);
        conv(upA, ilpw, ilpb, true, upD, ihpw, ihpb,
             true, ip2[pi], nullptr, true, 2 * ln, 2 * ln, 1, 17, 8);
        normalize_kernel<<<dim3(cdiv(1024 * 2 * ln, 256)), dim3(256), 0, stream>>>(
            (ushort_t*)(ws + ip2[pi]), stats, (float)((size_t)1024 * 2 * ln),
            0, 2 * ln);
        cur = ip2[pi]; pi ^= 1;
    }

    // --- final projection: 1x1 conv == GEMM, f32 output (b, n, d) ---
    conv(cur, outw, outb, false, 0, nullptr, nullptr,
         false, 0, (float*)d_out, false, N, N, 1, 1, 0);
}
